// SAGE_78417512891172
// MI455X (gfx1250) — matrix operand order repaired and checked by reading, not compiled, not run on hardware
//
#include <hip/hip_runtime.h>

// ---------------------------------------------------------------------------
// GraphSAGE 3-layer inference for MI455X (gfx1250, wave32, WMMA).
//   N = [1081344, 67584, 6144, 1024] nodes per level
//   E = [1013760, 61440, 5120] edges per level
//   feats: 100 -> 256 -> 256 -> 47
// Strategy:
//   * fused gather into edge scatter-add (never materialize the 432MB h)
//   * weights pre-packed per call into the exact v_wmma_f32_16x16x32_f16
//     B-fragment layout (f16, zero-padded for ragged K/N) -> B fragment is
//     one contiguous 32B load per lane, no guards, no cvt in the hot loop
//   * layer-0 A buffers padded to Kp=128 so the GEMM K-loop is guard-free
// ---------------------------------------------------------------------------

typedef __attribute__((ext_vector_type(16))) _Float16 v16h;
typedef __attribute__((ext_vector_type(8)))  float    v8f;

#define N0 1081344
#define N1 67584
#define N2 6144
#define N3 1024
#define E0 1013760
#define E1 61440
#define E2 5120
#define INF 100
#define HID 256
#define CLS 47

static inline int cdiv(long a, long b) { return (int)((a + b - 1) / b); }

// -------------------------------- zero fill --------------------------------
__global__ void zero_f32(float* __restrict__ p, long n) {
    long i = (long)blockIdx.x * blockDim.x + threadIdx.x;
    long stride = (long)gridDim.x * blockDim.x;
    for (; i < n; i += stride) p[i] = 0.0f;
}

// ---------------- gather rows with K padding: out[r, 0..Kp) ----------------
// out[r*Kp + f] = f < K ? emb[idx[r]*K + f] : 0
__global__ void gather_rows_pad(const int* __restrict__ idx,
                                const float* __restrict__ emb,
                                float* __restrict__ out,
                                long nrows, int K, int Kp) {
    long i = (long)blockIdx.x * blockDim.x + threadIdx.x;
    long total = nrows * (long)Kp;
    if (i >= total) return;
    long r = i / Kp;
    int  f = (int)(i - r * Kp);
    out[i] = (f < K) ? emb[(long)idx[r] * K + f] : 0.0f;
}

// --------- pack W[K,N] (f32, row-major) into WMMA B-fragment layout --------
// Fragment layout for v_wmma_f32_16x16x32_f16 (B is 32x16 per tile):
//   Wp[((kt*tilesN + nt)*32 + lane)*16 + h] = W[(kt*32 + (h&8)*2 + (lane>>4)*8 + (h&7))*N
//                                               + nt*16 + (lane&15)]
// zero-padded where kk >= K or col >= N.
__global__ void pack_w_f16(const float* __restrict__ W, _Float16* __restrict__ Wp,
                           int K, int N, int tilesN, long total) {
    long i = (long)blockIdx.x * blockDim.x + threadIdx.x;
    if (i >= total) return;
    int  h    = (int)(i & 15);
    int  lane = (int)((i >> 4) & 31);
    long tile = i >> 9;
    int  nt = (int)(tile % tilesN);
    int  kt = (int)(tile / tilesN);
    int  kk  = kt * 32 + (h & 8) * 2 + (lane >> 4) * 8 + (h & 7);
    int  col = nt * 16 + (lane & 15);
    _Float16 v = (_Float16)0.0f;
    if (kk < K && col < N) v = (_Float16)W[(long)kk * N + col];
    Wp[i] = v;
}

// ----------------- edge scatter-add (wave-per-edge, fused gather) ----------
// agg[dst*aggStride + f] += feat[(gather?gather[src]:src)*K + f];  cnt[dst]+=1
__global__ void edge_agg(const int* __restrict__ src, const int* __restrict__ dst,
                         const int* __restrict__ gather,
                         const float* __restrict__ feat, int K, int aggStride,
                         int E, float* __restrict__ agg, float* __restrict__ cnt) {
    int wavesPerBlock = blockDim.x >> 5;
    int wave = blockIdx.x * wavesPerBlock + (threadIdx.x >> 5);
    int lane = threadIdx.x & 31;
    if (wave >= E) return;
    int s = src[wave];
    int d = dst[wave];
    long srow = gather ? (long)gather[s] : (long)s;
    const float* frow = feat + srow * (long)K;
    float* arow = agg + (long)d * (long)aggStride;
    for (int f = lane; f < K; f += 32)
        atomicAdd(&arow[f], frow[f]);
    if (lane == 0)
        atomicAdd(&cnt[d], 1.0f);
}

// ----------------------------- WMMA SAGE GEMM ------------------------------
// out[M,N] = dstF @ Ws + (agg/max(cnt,1)) @ Wn + b   (A stride = Kp, padded)
// One wave per 16x16 output tile; two v_wmma_f32_16x16x32_f16 per K-step.
// Guard-free inner loop: A is Kp-padded, B is pre-packed/zero-padded f16.
__global__ void sage_gemm_wmma(const float* __restrict__ dstF,
                               const float* __restrict__ agg,
                               const float* __restrict__ cnt,
                               const _Float16* __restrict__ Wsp,
                               const _Float16* __restrict__ Wnp,
                               const float* __restrict__ bias,
                               float* __restrict__ out,
                               int Kp, int Ksteps, int N, int tilesN,
                               int totalTiles, int doRelu) {
    int wavesPerBlock = blockDim.x >> 5;
    int wtile = blockIdx.x * wavesPerBlock + (threadIdx.x >> 5);
    if (wtile >= totalTiles) return;           // wave-uniform: EXEC stays full
    int lane   = threadIdx.x & 31;
    int laneLo = lane & 15;
    int laneHi = lane >> 4;

    int tile_m = wtile / tilesN;
    int tile_n = wtile - tile_m * tilesN;

    int row = tile_m * 16 + laneLo;            // A-fragment row
    float invc = 1.0f / fmaxf(cnt[row], 1.0f);
    const float* drow = dstF + (long)row * Kp;
    const float* arow = agg  + (long)row * Kp;

    // Packed-B base for this wave: fragment tile (kt, tile_n), this lane.
    const _Float16* pbs = Wsp + ((long)tile_n * 32 + lane) * 16;
    const _Float16* pbn = Wnp + ((long)tile_n * 32 + lane) * 16;
    long bstep = (long)tilesN * 32 * 16;       // halves per K-step

    v8f c = {};
    for (int kt = 0; kt < Ksteps; ++kt) {
        int k0 = kt * 32;
        // ---- A fragments: two contiguous 8-float runs per matrix --------
        // a[0..7]  = row[k0 + laneHi*8 + 0..7]
        // a[8..15] = row[k0 + 16 + laneHi*8 + 0..7]
        v16h a_d, a_n;
        {
            const float4* p0 = (const float4*)(drow + k0 + laneHi * 8);
            const float4* p1 = (const float4*)(drow + k0 + 16 + laneHi * 8);
            const float4* q0 = (const float4*)(arow + k0 + laneHi * 8);
            const float4* q1 = (const float4*)(arow + k0 + 16 + laneHi * 8);
            float4 d0 = p0[0], d1 = p0[1], d2 = p1[0], d3 = p1[1];
            float4 n0 = q0[0], n1 = q0[1], n2 = q1[0], n3 = q1[1];
            a_d[0]=(_Float16)d0.x; a_d[1]=(_Float16)d0.y; a_d[2]=(_Float16)d0.z; a_d[3]=(_Float16)d0.w;
            a_d[4]=(_Float16)d1.x; a_d[5]=(_Float16)d1.y; a_d[6]=(_Float16)d1.z; a_d[7]=(_Float16)d1.w;
            a_d[8]=(_Float16)d2.x; a_d[9]=(_Float16)d2.y; a_d[10]=(_Float16)d2.z; a_d[11]=(_Float16)d2.w;
            a_d[12]=(_Float16)d3.x; a_d[13]=(_Float16)d3.y; a_d[14]=(_Float16)d3.z; a_d[15]=(_Float16)d3.w;
            a_n[0]=(_Float16)(n0.x*invc); a_n[1]=(_Float16)(n0.y*invc); a_n[2]=(_Float16)(n0.z*invc); a_n[3]=(_Float16)(n0.w*invc);
            a_n[4]=(_Float16)(n1.x*invc); a_n[5]=(_Float16)(n1.y*invc); a_n[6]=(_Float16)(n1.z*invc); a_n[7]=(_Float16)(n1.w*invc);
            a_n[8]=(_Float16)(n2.x*invc); a_n[9]=(_Float16)(n2.y*invc); a_n[10]=(_Float16)(n2.z*invc); a_n[11]=(_Float16)(n2.w*invc);
            a_n[12]=(_Float16)(n3.x*invc); a_n[13]=(_Float16)(n3.y*invc); a_n[14]=(_Float16)(n3.z*invc); a_n[15]=(_Float16)(n3.w*invc);
        }
        // ---- B fragments: contiguous 32B per lane, pre-packed f16 -------
        v16h b_s = *(const v16h*)(pbs + (long)kt * bstep);
        v16h b_n = *(const v16h*)(pbn + (long)kt * bstep);

        c = __builtin_amdgcn_wmma_f32_16x16x32_f16(false, a_d, false, b_s,
                                                   (short)0, c, false, false);
        c = __builtin_amdgcn_wmma_f32_16x16x32_f16(false, a_n, false, b_n,
                                                   (short)0, c, false, false);
    }

    // Epilogue. C/D layout: VGPR r -> M = r + 8*laneHi ; N = laneLo.
    int n = tile_n * 16 + laneLo;
    if (n < N) {
        float bb = bias[n];
        #pragma unroll
        for (int r = 0; r < 8; ++r) {
            int m = tile_m * 16 + r + laneHi * 8;
            float v = c[r] + bb;
            if (doRelu) v = fmaxf(v, 0.0f);
            out[(long)m * N + n] = v;
        }
    }
}

// --------------------------------- driver ----------------------------------
extern "C" void kernel_launch(void* const* d_in, const int* in_sizes, int n_in,
                              void* d_out, int out_size, void* d_ws, size_t ws_size,
                              hipStream_t stream) {
    (void)in_sizes; (void)n_in; (void)out_size; (void)ws_size;
    const int*   input_nodes = (const int*)  d_in[0];
    const int*   src0 = (const int*)  d_in[1];
    const int*   dst0 = (const int*)  d_in[2];
    const int*   src1 = (const int*)  d_in[3];
    const int*   dst1 = (const int*)  d_in[4];
    const int*   src2 = (const int*)  d_in[5];
    const int*   dst2 = (const int*)  d_in[6];
    const float* emb  = (const float*)d_in[7];
    const float* Wn0  = (const float*)d_in[8];
    const float* Ws0  = (const float*)d_in[9];
    const float* b0   = (const float*)d_in[10];
    const float* Wn1  = (const float*)d_in[11];
    const float* Ws1  = (const float*)d_in[12];
    const float* b1   = (const float*)d_in[13];
    const float* Wn2  = (const float*)d_in[14];
    const float* Ws2  = (const float*)d_in[15];
    const float* b2   = (const float*)d_in[16];
    float* out = (float*)d_out;

    const int KP0 = 128;                     // layer-0 K padded to 32-multiple
    const int TN0 = HID / 16, KS0 = KP0 / 32;          // 16 tiles, 4 k-steps
    const int TN1 = HID / 16, KS1 = HID / 32;          // 16 tiles, 8 k-steps
    const int TN2 = cdiv(CLS, 16), KS2 = HID / 32;     // 3 tiles, 8 k-steps
    const long PW0 = (long)KS0 * TN0 * 32 * 16;        // packed halves, layer 0
    const long PW1 = (long)KS1 * TN1 * 32 * 16;
    const long PW2 = (long)KS2 * TN2 * 32 * 16;

    // Workspace layout (floats). Aggregation chunk first: one zero pass.
    float* ws = (float*)d_ws;
    long off = 0;
    float* agg0 = ws + off; off += (long)N1 * KP0;     // padded stride
    float* cnt0 = ws + off; off += N1;
    float* agg1 = ws + off; off += (long)N2 * HID;
    float* cnt1 = ws + off; off += N2;
    float* agg2 = ws + off; off += (long)N3 * HID;
    float* cnt2 = ws + off; off += N3;
    long zeroElems = off;                              // contiguous agg/cnt
    float* hdst0 = ws + off; off += (long)N1 * KP0;    // padded self feats
    float* h1    = ws + off; off += (long)N1 * HID;
    float* h2    = ws + off; off += (long)N2 * HID;
    _Float16* Wsp0 = (_Float16*)(ws + off); off += PW0 / 2;
    _Float16* Wnp0 = (_Float16*)(ws + off); off += PW0 / 2;
    _Float16* Wsp1 = (_Float16*)(ws + off); off += PW1 / 2;
    _Float16* Wnp1 = (_Float16*)(ws + off); off += PW1 / 2;
    _Float16* Wsp2 = (_Float16*)(ws + off); off += PW2 / 2;
    _Float16* Wnp2 = (_Float16*)(ws + off); off += PW2 / 2;

    // 1) zero aggregation buffers (re-zeroed every call: deterministic).
    zero_f32<<<2048, 256, 0, stream>>>(ws, zeroElems);

    // 2) pack weights into WMMA B-fragment layout (tiny, per call).
    pack_w_f16<<<cdiv(PW0, 256), 256, 0, stream>>>(Ws0, Wsp0, INF, HID, TN0, PW0);
    pack_w_f16<<<cdiv(PW0, 256), 256, 0, stream>>>(Wn0, Wnp0, INF, HID, TN0, PW0);
    pack_w_f16<<<cdiv(PW1, 256), 256, 0, stream>>>(Ws1, Wsp1, HID, HID, TN1, PW1);
    pack_w_f16<<<cdiv(PW1, 256), 256, 0, stream>>>(Wn1, Wnp1, HID, HID, TN1, PW1);
    pack_w_f16<<<cdiv(PW2, 256), 256, 0, stream>>>(Ws2, Wsp2, HID, CLS, TN2, PW2);
    pack_w_f16<<<cdiv(PW2, 256), 256, 0, stream>>>(Wn2, Wnp2, HID, CLS, TN2, PW2);

    // 3) gather self features for layer-0 destination nodes (zero-padded).
    gather_rows_pad<<<cdiv((long)N1 * KP0, 256), 256, 0, stream>>>(
        input_nodes, emb, hdst0, N1, INF, KP0);

    // 4) layer-0 neighbor aggregation (fused input_nodes gather, wave/edge).
    edge_agg<<<cdiv(E0, 8), 256, 0, stream>>>(src0, dst0, input_nodes, emb,
                                              INF, KP0, E0, agg0, cnt0);

    // 5) layer-0 GEMM + bias + relu  -> h1 [N1, 256]
    {
        int total = (N1 / 16) * TN0;
        sage_gemm_wmma<<<cdiv(total, 4), 128, 0, stream>>>(
            hdst0, agg0, cnt0, Wsp0, Wnp0, b0, h1, KP0, KS0, HID, TN0, total, 1);
    }

    // 6) layer-1 aggregation over h1.
    edge_agg<<<cdiv(E1, 8), 256, 0, stream>>>(src1, dst1, nullptr, h1,
                                              HID, HID, E1, agg1, cnt1);

    // 7) layer-1 GEMM + bias + relu  -> h2 [N2, 256]  (self = h1 rows 0..N2)
    {
        int total = (N2 / 16) * TN1;
        sage_gemm_wmma<<<cdiv(total, 4), 128, 0, stream>>>(
            h1, agg1, cnt1, Wsp1, Wnp1, b1, h2, HID, KS1, HID, TN1, total, 1);
    }

    // 8) layer-2 aggregation over h2.
    edge_agg<<<cdiv(E2, 8), 256, 0, stream>>>(src2, dst2, nullptr, h2,
                                              HID, HID, E2, agg2, cnt2);

    // 9) layer-2 GEMM + bias (no relu) -> d_out [N3, 47]
    {
        int total = (N3 / 16) * TN2;
        sage_gemm_wmma<<<cdiv(total, 4), 128, 0, stream>>>(
            h2, agg2, cnt2, Wsp2, Wnp2, b2, out, HID, KS2, CLS, TN2, total, 0);
    }
}
